// GCNNet2_51599737094934
// MI455X (gfx1250) — matrix operand order repaired and checked by reading, not compiled, not run on hardware
//
#include <hip/hip_runtime.h>
#include <hip/hip_bf16.h>
#include <stdint.h>
#include <stddef.h>

// ---------------- problem constants ----------------
#define NN    50000
#define EE    800000
#define DINF  146
#define DD    128
#define GGR   64
#define LLAY  4
#define NCLS  10
#define EPSBN 1e-5f
#define MPAD  50048            // 50000 padded up to multiple of 128
#define KEMB  160              // 146 padded to multiple of 32
#define ENTOT (EE + NN)        // edges + self loops
#define LDSP  40               // LDS row pitch in halves (32 data + 8 pad)

typedef __attribute__((ext_vector_type(16))) _Float16 v16h;
typedef __attribute__((ext_vector_type(8)))  float    v8f;
typedef __attribute__((ext_vector_type(4)))  unsigned u32x4;
typedef __attribute__((ext_vector_type(8)))  int      i32x8;
typedef __attribute__((ext_vector_type(4)))  int      i32x4;

#if defined(__has_builtin)
#  if __has_builtin(__builtin_amdgcn_tensor_load_to_lds)
#    define HAVE_TDM 1
#  endif
#endif
#ifndef HAVE_TDM
#  define HAVE_TDM 0
#endif

union FragU { v16h v; uint4 q[2]; };

// ---- TDM: DMA one [128 rows x 32 halves] tile (row stride ldk halves) into
// ---- LDS with 4-DWORD padding every 16 DWORDs (-> 40-half LDS pitch).
__device__ inline void tdm_load_tile(const _Float16* g, int ldk, _Float16* lds) {
#if HAVE_TDM
  unsigned long long ga = (unsigned long long)(size_t)g;
  u32x4 g0;
  g0.x = 1u;                                             // count=1, user mode
  g0.y = (unsigned)(size_t)lds;                          // lds_addr (bytes)
  g0.z = (unsigned)ga;                                   // global_addr[31:0]
  g0.w = (unsigned)((ga >> 32) & 0x01FFFFFFull) | (2u << 30); // addr[56:32] | type=2
  i32x8 g1;
  unsigned td0 = (unsigned)ldk;
  g1[0] = (int)((1u << 16)        // data_size = 2 bytes
              | (1u << 20)        // pad_enable
              | (3u << 22)        // pad_interval: 16 DWORDs
              | (3u << 25));      // pad_amount: 4 DWORDs
  g1[1] = (int)((td0 & 0xFFFFu) << 16);                   // tensor_dim0[15:0]
  g1[2] = (int)(((td0 >> 16) & 0xFFFFu) | (60000u << 16));// tensor_dim0 hi | tensor_dim1 lo
  g1[3] = (int)(32u << 16);                               // tensor_dim1 hi=0 | tile_dim0=32
  g1[4] = 128;                                            // tile_dim1=128, tile_dim2=0
  g1[5] = (int)td0;                                       // tensor_dim0_stride[31:0]
  g1[6] = 0;
  g1[7] = 0;
  i32x4 z4 = (i32x4)0;
  i32x8 z8 = (i32x8)0;
  __builtin_amdgcn_tensor_load_to_lds(g0, g1, z4, z4, z8, 0);
#else
  (void)g; (void)ldk; (void)lds;
#endif
}

// ---- cooperative fallback loader: 256 threads, 128x32-half tile ----
__device__ inline void coop_load(const _Float16* g, int ldk, _Float16* lds, int t) {
  int row = t >> 1, part = t & 1;
  const uint4* s = (const uint4*)(g + (size_t)row * ldk + part * 16);
  uint4* d = (uint4*)(lds + row * LDSP + part * 16);
  d[0] = s[0];
  d[1] = s[1];
}

// ================= WMMA GEMM: C[M x 128] = A[M x K] * B^T ================
// A: f16 row-major [MPAD x K];  Bw: f16 pre-transposed [128 x K] (row n holds col n of W)
__global__ __launch_bounds__(256) void k_gemm(
    const _Float16* __restrict__ A, int K,
    const _Float16* __restrict__ Bw,
    const float* __restrict__ bias,
    float* __restrict__ C, _Float16* __restrict__ Ch) {
  __shared__ alignas(16) _Float16 ldsA[128 * LDSP];
  __shared__ alignas(16) _Float16 ldsB[128 * LDSP];
  const int t = threadIdx.x;
  const int lane = t & 31, w = t >> 5;
  const int wm = (w & 3) * 32;        // wave M offset
  const int wn = (w >> 2) * 64;       // wave N offset
  const int lm = lane & 15, hi = lane >> 4;
  const int r0 = blockIdx.x * 128;
  v8f acc[2][4] = {};

  const int nsteps = K / 32;
  for (int ks = 0; ks < nsteps; ++ks) {
#if HAVE_TDM
    if (t < 32) {                               // one wave issues both DMAs
      tdm_load_tile(A + (size_t)r0 * K + ks * 32, K, ldsA);
      tdm_load_tile(Bw + ks * 32, K, ldsB);
      __builtin_amdgcn_s_wait_tensorcnt(0);
    }
#else
    coop_load(A + (size_t)r0 * K + ks * 32, K, ldsA, t);
    coop_load(Bw + ks * 32, K, ldsB, t);
#endif
    __syncthreads();

    FragU a[2], b[4];
#pragma unroll
    for (int mi = 0; mi < 2; ++mi) {
      int row = wm + mi * 16 + lm;
      a[mi].q[0] = *(const uint4*)(ldsA + row * LDSP + hi * 8);       // K = 8hi..8hi+7
      a[mi].q[1] = *(const uint4*)(ldsA + row * LDSP + 16 + hi * 8);  // K = 16+8hi..+7
    }
#pragma unroll
    for (int ni = 0; ni < 4; ++ni) {
      int col = wn + ni * 16 + lm;
      b[ni].q[0] = *(const uint4*)(ldsB + col * LDSP + hi * 8);       // K = 8hi..8hi+7
      b[ni].q[1] = *(const uint4*)(ldsB + col * LDSP + 16 + hi * 8);  // K = 16+8hi..+7
    }
#pragma unroll
    for (int mi = 0; mi < 2; ++mi)
#pragma unroll
      for (int ni = 0; ni < 4; ++ni)
        acc[mi][ni] = __builtin_amdgcn_wmma_f32_16x16x32_f16(
            false, a[mi].v, false, b[ni].v, (short)0, acc[mi][ni], false, false);
    __syncthreads();
  }

#pragma unroll
  for (int mi = 0; mi < 2; ++mi) {
#pragma unroll
    for (int ni = 0; ni < 4; ++ni) {
      const int col = wn + ni * 16 + lm;
      const float bv = bias ? bias[col] : 0.0f;
#pragma unroll
      for (int v = 0; v < 8; ++v) {
        const int row = r0 + wm + mi * 16 + hi * 8 + v;
        const float val = acc[mi][ni][v] + bv;
        C[(size_t)row * DD + col] = val;
        if (Ch) Ch[(size_t)row * DD + col] = (_Float16)val;
      }
    }
  }
}

// ================= graph prep =================
__global__ void k_deg_init(float* d) {
  int i = blockIdx.x * 256 + threadIdx.x;
  if (i < NN) d[i] = 1.0f;                    // self-loop
}
__global__ void k_deg_acc(const int* __restrict__ ei, float* d) {
  int e = blockIdx.x * 256 + threadIdx.x;
  if (e < EE) atomicAdd(&d[ei[EE + e]], 1.0f);
}
__global__ void k_deg_fin(float* d) {
  int i = blockIdx.x * 256 + threadIdx.x;
  if (i < NN) d[i] = rsqrtf(d[i]);
}

// ================= converters =================
__global__ void k_cvt_x(const float* __restrict__ x, _Float16* __restrict__ xh) {
  size_t idx = (size_t)blockIdx.x * 256 + threadIdx.x;
  if (idx >= (size_t)MPAD * KEMB) return;
  int i = (int)(idx / KEMB), k = (int)(idx % KEMB);
  xh[idx] = (i < NN && k < DINF) ? (_Float16)x[(size_t)i * DINF + k] : (_Float16)0.0f;
}
__global__ void k_cvt_w(const float* __restrict__ Wemb, const float* __restrict__ Wg,
                        _Float16* __restrict__ wT) {
  int idx = blockIdx.x * 256 + threadIdx.x;
  if (idx < KEMB * DD) {                               // W_emb^T padded: [128][160]
    int n = idx / KEMB, k = idx % KEMB;
    wT[idx] = (k < DINF) ? (_Float16)Wemb[(size_t)k * DD + n] : (_Float16)0.0f;
  } else if (idx < KEMB * DD + LLAY * DD * DD) {       // W_gcn^T: 4 x [128][128]
    int r = idx - KEMB * DD;
    int l = r / (DD * DD), q = r % (DD * DD);
    int n = q / DD, k = q % DD;
    wT[idx] = (_Float16)Wg[((size_t)l * DD + k) * DD + n];
  }
}

// ================= per-layer ops =================
__global__ void k_zero(float* p, long n) {
  long i = (long)blockIdx.x * 256 + threadIdx.x;
  if (i < n) p[i] = 0.0f;
}
__global__ void k_agg_init(float* __restrict__ agg, const float* __restrict__ b) {
  size_t idx = (size_t)blockIdx.x * 256 + threadIdx.x;
  if (idx < (size_t)NN * DD) agg[idx] = b[idx & (DD - 1)];
}
// one wave per edge: agg[dst] += m[src] * (dinv[src]*dinv[dst])
__global__ void k_scatter(const float* __restrict__ m, const int* __restrict__ ei,
                          const float* __restrict__ dinv, float* __restrict__ agg) {
  int wid = threadIdx.x >> 5, lane = threadIdx.x & 31;
  long e = (long)blockIdx.x * 8 + wid;
  if (e >= ENTOT) return;
  int s, d;
  if (e < EE) { s = ei[e]; d = ei[EE + e]; }
  else        { s = d = (int)(e - EE); }
  float nrm = dinv[s] * dinv[d];
  float4 v = *(const float4*)(m + (size_t)s * DD + lane * 4);
  float* ap = agg + (size_t)d * DD + lane * 4;
  atomicAdd(ap + 0, v.x * nrm);
  atomicAdd(ap + 1, v.y * nrm);
  atomicAdd(ap + 2, v.z * nrm);
  atomicAdd(ap + 3, v.w * nrm);
}
// column sums / sums-of-squares (grid=128 blocks)
__global__ void k_bnstats(const float* __restrict__ agg, float* __restrict__ bn) {
  __shared__ float sm[512];
  int t = threadIdx.x, j = t & 127, rg = t >> 7;
  float s = 0.0f, ss = 0.0f;
  for (int i = blockIdx.x * 2 + rg; i < NN; i += 256) {
    float v = agg[(size_t)i * DD + j];
    s += v; ss += v * v;
  }
  sm[t] = s; sm[256 + t] = ss;
  __syncthreads();
  if (rg == 0) {
    atomicAdd(&bn[j],      sm[t] + sm[t + 128]);
    atomicAdd(&bn[DD + j], sm[256 + t] + sm[256 + t + 128]);
  }
}
// fused BN(batch stats) + ReLU + residual + f16 recast; zeros the M-pad rows
__global__ void k_apply(const float* __restrict__ agg, const float* __restrict__ bn,
                        const float* __restrict__ gam, const float* __restrict__ bet,
                        float* __restrict__ h, _Float16* __restrict__ hf) {
  size_t idx = (size_t)blockIdx.x * 256 + threadIdx.x;
  if (idx >= (size_t)MPAD * DD) return;
  int i = (int)(idx >> 7), j = (int)(idx & 127);
  float y = 0.0f;
  if (i < NN) {
    float a  = agg[idx];
    float mu = bn[j] * (1.0f / NN);
    float var = bn[DD + j] * (1.0f / NN) - mu * mu;
    float tv = (a - mu) * rsqrtf(var + EPSBN) * gam[j] + bet[j];
    y = fmaxf(tv, 0.0f) + h[idx];
  }
  h[idx] = y;
  hf[idx] = (_Float16)y;
}

// ================= readout =================
__global__ void k_pool(const float* __restrict__ h, const int* __restrict__ batch,
                       float* __restrict__ pool, float* __restrict__ cnt) {
  size_t idx = (size_t)blockIdx.x * 256 + threadIdx.x;
  if (idx >= (size_t)NN * DD) return;
  int i = (int)(idx >> 7), j = (int)(idx & 127);
  int b = batch[i];
  atomicAdd(&pool[(size_t)b * DD + j], h[idx]);
  if (j == 0) atomicAdd(&cnt[b], 1.0f);
}
__global__ __launch_bounds__(256) void k_readout(
    const float* __restrict__ pool, const float* __restrict__ cnt,
    const float* __restrict__ W1, const float* __restrict__ b1,
    const float* __restrict__ W2, const float* __restrict__ b2,
    const float* __restrict__ W3, const float* __restrict__ b3,
    float* __restrict__ out) {
  __shared__ float hg[GGR * DD];
  int t = threadIdx.x;
  for (int idx = t; idx < GGR * DD; idx += 256) {
    int g = idx >> 7;
    float c = cnt[g]; c = c < 1.0f ? 1.0f : c;
    hg[idx] = pool[idx] / c;
  }
  __syncthreads();
  if (t < GGR) {
    float z1[64];
    for (int o = 0; o < 64; ++o) {
      float s = b1[o];
      for (int i = 0; i < 128; ++i) s += hg[t * DD + i] * W1[i * 64 + o];
      z1[o] = fmaxf(s, 0.0f);
    }
    float z2[32];
    for (int o = 0; o < 32; ++o) {
      float s = b2[o];
      for (int i = 0; i < 64; ++i) s += z1[i] * W2[i * 32 + o];
      z2[o] = fmaxf(s, 0.0f);
    }
    for (int o = 0; o < NCLS; ++o) {
      float s = b3[o];
      for (int i = 0; i < 32; ++i) s += z2[i] * W3[i * NCLS + o];
      out[t * NCLS + o] = s;
    }
  }
}

// ================= host launcher =================
extern "C" void kernel_launch(void* const* d_in, const int* in_sizes, int n_in,
                              void* d_out, int out_size, void* d_ws, size_t ws_size,
                              hipStream_t stream) {
  (void)in_sizes; (void)n_in; (void)out_size; (void)ws_size;
  const float* x    = (const float*)d_in[0];
  const int*   ei   = (const int*)d_in[1];
  const int*   batc = (const int*)d_in[2];
  const float* Wemb = (const float*)d_in[3];
  const float* bemb = (const float*)d_in[4];
  const float* Wg   = (const float*)d_in[5];
  const float* bg   = (const float*)d_in[6];
  const float* gam  = (const float*)d_in[7];
  const float* bet  = (const float*)d_in[8];
  const float* W1   = (const float*)d_in[9];
  const float* b1   = (const float*)d_in[10];
  const float* W2   = (const float*)d_in[11];
  const float* b2   = (const float*)d_in[12];
  const float* W3   = (const float*)d_in[13];
  const float* b3   = (const float*)d_in[14];
  float* out = (float*)d_out;

  // workspace carve
  size_t off = 0;
  char* base = (char*)d_ws;
  auto carve = [&](size_t bytes) -> void* {
    void* p = base + off;
    off += (bytes + 255) & ~(size_t)255;
    return p;
  };
  float*     dinv = (float*)carve((size_t)NN * 4);
  _Float16*  xh   = (_Float16*)carve((size_t)MPAD * KEMB * 2);
  _Float16*  wT   = (_Float16*)carve((size_t)(KEMB * DD + LLAY * DD * DD) * 2);
  float*     h    = (float*)carve((size_t)MPAD * DD * 4);
  _Float16*  hf   = (_Float16*)carve((size_t)MPAD * DD * 2);
  float*     m    = (float*)carve((size_t)MPAD * DD * 4);
  float*     agg  = (float*)carve((size_t)MPAD * DD * 4);
  float*     bn   = (float*)carve(256 * 4);
  float*     pool = (float*)carve((size_t)GGR * DD * 4);
  float*     cntb = (float*)carve((size_t)GGR * 4);

  auto cdiv = [](long a, long b) { return (int)((a + b - 1) / b); };

  // degree -> dinv
  k_deg_init<<<cdiv(NN, 256), 256, 0, stream>>>(dinv);
  k_deg_acc<<<cdiv(EE, 256), 256, 0, stream>>>(ei, dinv);
  k_deg_fin<<<cdiv(NN, 256), 256, 0, stream>>>(dinv);

  // f16 conversions (x padded to [MPAD x 160], weights transposed [N][K])
  k_cvt_x<<<cdiv((long)MPAD * KEMB, 256), 256, 0, stream>>>(x, xh);
  k_cvt_w<<<cdiv(KEMB * DD + LLAY * DD * DD, 256), 256, 0, stream>>>(Wemb, Wg, wT);

  // embedding: h = x @ W_emb + b_emb   (writes f32 + f16 copies)
  k_gemm<<<MPAD / 128, 256, 0, stream>>>(xh, KEMB, wT, bemb, h, hf);

  for (int l = 0; l < LLAY; ++l) {
    k_agg_init<<<cdiv((long)NN * DD, 256), 256, 0, stream>>>(agg, bg + l * DD);
    k_gemm<<<MPAD / 128, 256, 0, stream>>>(hf, DD, wT + KEMB * DD + l * DD * DD,
                                           (const float*)nullptr, m, (_Float16*)nullptr);
    k_scatter<<<cdiv(ENTOT, 8), 256, 0, stream>>>(m, ei, dinv, agg);
    k_zero<<<1, 256, 0, stream>>>(bn, 256);
    k_bnstats<<<128, 256, 0, stream>>>(agg, bn);
    k_apply<<<cdiv((long)MPAD * DD, 256), 256, 0, stream>>>(agg, bn, gam + l * DD,
                                                            bet + l * DD, h, hf);
  }

  k_zero<<<1, 256, 0, stream>>>(cntb, GGR);
  k_zero<<<cdiv(GGR * DD, 256), 256, 0, stream>>>(pool, GGR * DD);
  k_pool<<<cdiv((long)NN * DD, 256), 256, 0, stream>>>(h, batc, pool, cntb);
  k_readout<<<1, 256, 0, stream>>>(pool, cntb, W1, b1, W2, b2, W3, b3, out);
}
